// _MoEBlockWithGateRouter_45277545234436
// MI455X (gfx1250) — matrix operand order repaired and checked by reading, not compiled, not run on hardware
//
#include <hip/hip_runtime.h>
#include <cstdint>
#include <cstddef>

#define T_TOKENS 16384
#define DM       2048
#define NEXP     8

#define BM 128
#define BN 128
#define BK 32
#define LDPF 36   // f32 LDS pitch: 144B (16B-aligned float4 reads, 16-bank spread)
#define KITERS (DM / BK)   // 64

typedef __attribute__((ext_vector_type(16))) _Float16 v16h;
typedef __attribute__((ext_vector_type(8)))  float    v8f;

// ---------------------------------------------------------------------------
// gfx1250 async copy: 16B/lane global -> LDS, tracked by ASYNCcnt (no VGPR dst)
// ---------------------------------------------------------------------------
__device__ __forceinline__ void async_copy_b128(void* lds_ptr, const float* gptr) {
  const uint32_t lds_off = (uint32_t)(uintptr_t)lds_ptr;  // low 32b of flat = LDS offset
  asm volatile("global_load_async_to_lds_b128 %0, %1, off"
               :: "v"(lds_off), "v"((uint64_t)(uintptr_t)gptr)
               : "memory");
}
__device__ __forceinline__ void wait_async0() {
  asm volatile("s_wait_asynccnt 0" ::: "memory");
}

__device__ __forceinline__ v16h make_frag(const float4 q0, const float4 q1,
                                          const float4 q2, const float4 q3) {
  v16h r = { (_Float16)q0.x, (_Float16)q0.y, (_Float16)q0.z, (_Float16)q0.w,
             (_Float16)q1.x, (_Float16)q1.y, (_Float16)q1.z, (_Float16)q1.w,
             (_Float16)q2.x, (_Float16)q2.y, (_Float16)q2.z, (_Float16)q2.w,
             (_Float16)q3.x, (_Float16)q3.y, (_Float16)q3.z, (_Float16)q3.w };
  return r;
}

// ---------------------------------------------------------------------------
// Stage 1: router — logits = H @ Wg^T + bg, top-2 per token, bucket by expert
// ---------------------------------------------------------------------------
__global__ __launch_bounds__(256) void moe_router_kernel(
    const float* __restrict__ H, const float* __restrict__ Wg,
    const float* __restrict__ bg, int* __restrict__ cnt, int* __restrict__ lists)
{
  __shared__ float sWg[NEXP * DM];   // 64 KB
  const int tid = threadIdx.x;
  #pragma unroll
  for (int j = 0; j < 16; ++j) {
    const int idx4 = tid + j * 256;
    *(float4*)&sWg[idx4 * 4] = *(const float4*)&Wg[idx4 * 4];
  }
  __syncthreads();

  const int lane = tid & 31;
  const int wid  = tid >> 5;
  const int wavesTotal = gridDim.x * 8;

  for (int tok = blockIdx.x * 8 + wid; tok < T_TOKENS; tok += wavesTotal) {
    float acc[NEXP];
    #pragma unroll
    for (int e = 0; e < NEXP; ++e) acc[e] = 0.0f;

    const float4* hrow = (const float4*)(H + (size_t)tok * DM);
    for (int it = 0; it < 16; ++it) {
      const int k4 = lane + it * 32;
      const float4 h = hrow[k4];
      #pragma unroll
      for (int e = 0; e < NEXP; ++e) {
        const float4 w = *(const float4*)&sWg[e * DM + k4 * 4];
        acc[e] += h.x * w.x + h.y * w.y + h.z * w.z + h.w * w.w;
      }
    }
    #pragma unroll
    for (int e = 0; e < NEXP; ++e) {
      #pragma unroll
      for (int off = 16; off > 0; off >>= 1)
        acc[e] += __shfl_xor(acc[e], off, 32);
    }
    if (lane == 0) {
      float lg[NEXP];
      #pragma unroll
      for (int e = 0; e < NEXP; ++e) lg[e] = acc[e] + bg[e];
      int e1 = 0; float v1 = lg[0];
      #pragma unroll
      for (int e = 1; e < NEXP; ++e) if (lg[e] > v1) { v1 = lg[e]; e1 = e; }
      int e2 = -1; float v2 = 0.0f;
      #pragma unroll
      for (int e = 0; e < NEXP; ++e) {
        if (e == e1) continue;
        if (e2 < 0 || lg[e] > v2) { v2 = lg[e]; e2 = e; }
      }
      int p1 = atomicAdd(&cnt[e1], 1); lists[e1 * T_TOKENS + p1] = tok;
      int p2 = atomicAdd(&cnt[e2], 1); lists[e2 * T_TOKENS + p2] = tok;
    }
  }
}

// ---------------------------------------------------------------------------
// Stage 2: gathered expert GEMM: out[tok] += gather(H) @ W[e]^T + b[e]
// f32 tiles DMA'd global->LDS via async ops (double-buffered), f16 conversion
// fused into fragment assembly, v_wmma_f32_16x16x32_f16 with f32 accumulate.
// sched_barrier(0) fences: (a) pin async DMA issue at iteration top,
// (b) materialize all 6 fragments before WMMA issue -> back-to-back WMMAs,
// single hazard window, coarse dscnt waits.
// ---------------------------------------------------------------------------
__global__ __launch_bounds__(256, 1) void moe_gemm_kernel(
    const float* __restrict__ H, const float* __restrict__ W,
    const float* __restrict__ Bexp, float* __restrict__ out,
    const int* __restrict__ cnt, const int* __restrict__ lists)
{
  const int e  = blockIdx.z;
  const int n0 = blockIdx.y * BN;
  const int m0 = blockIdx.x * BM;
  const int count = cnt[e];
  if (m0 >= count) return;                 // uniform early-exit (no barriers yet)

  __shared__ float fA[2][BM * LDPF];       // 2 x 18 KB (f32 tiles)
  __shared__ float fB[2][BN * LDPF];       // 2 x 18 KB
  __shared__ int   sTok[BM];

  const int tid = threadIdx.x;
  if (tid < BM) {
    int m = m0 + tid;
    if (m >= count) m = count - 1;         // clamp: rows >= count are never stored
    sTok[tid] = lists[e * T_TOKENS + m];
  }
  __syncthreads();

  const int lane  = tid & 31;
  const int wid   = tid >> 5;
  const int wm    = wid >> 2;              // 0..1  (M, 64 rows per wave)
  const int wn    = wid & 3;               // 0..3  (N, 32 cols per wave)
  const int lhalf = lane >> 4;             // 0/1
  const int lmod  = lane & 15;

  const float* Wbase = W + (size_t)e * DM * DM;

  // accumulators initialized with the expert bias (col-dependent only)
  v8f acc[4][2];
  #pragma unroll
  for (int ni = 0; ni < 2; ++ni) {
    const float bv = Bexp[(size_t)e * DM + n0 + wn * 32 + ni * 16 + lmod];
    #pragma unroll
    for (int mi = 0; mi < 4; ++mi) {
      v8f z = {};
      acc[mi][ni] = z + bv;
    }
  }

  // staging map: 128 rows x 8 float4-cols (BK=32 floats); 4 rows per thread
  const int c4  = tid & 7;                 // float4 column (k/4)
  const int rb0 = tid >> 3;                // 0..31; rows rb0 + i*32
  const int tokRow[4] = { sTok[rb0], sTok[rb0 + 32], sTok[rb0 + 64], sTok[rb0 + 96] };

  // prologue: DMA tile k0=0 into buffer 0
  #pragma unroll
  for (int i = 0; i < 4; ++i) {
    const int row = rb0 + i * 32;
    async_copy_b128(&fA[0][row * LDPF + c4 * 4], H + (size_t)tokRow[i] * DM + c4 * 4);
    async_copy_b128(&fB[0][row * LDPF + c4 * 4], Wbase + (size_t)(n0 + row) * DM + c4 * 4);
  }
  wait_async0();
  __syncthreads();

  for (int it = 0; it < KITERS; ++it) {
    const int cur = it & 1;
    const bool more = (it + 1) < KITERS;

    // (1) kick off next tile's async DMA into the other buffer (no VGPR dst,
    //     no wait here -> overlaps fully with the WMMA phase below)
    if (more) {
      const int kn = (it + 1) * BK;
      #pragma unroll
      for (int i = 0; i < 4; ++i) {
        const int row = rb0 + i * 32;
        async_copy_b128(&fA[cur ^ 1][row * LDPF + c4 * 4],
                        H + (size_t)tokRow[i] * DM + kn + c4 * 4);
        async_copy_b128(&fB[cur ^ 1][row * LDPF + c4 * 4],
                        Wbase + (size_t)(n0 + row) * DM + kn + c4 * 4);
      }
    }
    __builtin_amdgcn_sched_barrier(0);     // keep DMA issue at iteration top

    // (2) fragment assembly (LDS f32 -> v_cvt_pk_f16_f32 -> v16h)
    v16h a[4];
    #pragma unroll
    for (int mi = 0; mi < 4; ++mi) {
      const float* base = &fA[cur][(wm * 64 + mi * 16 + lmod) * LDPF];
      const float4 q0 = *(const float4*)&base[lhalf * 8];
      const float4 q1 = *(const float4*)&base[lhalf * 8 + 4];
      const float4 q2 = *(const float4*)&base[16 + lhalf * 8];
      const float4 q3 = *(const float4*)&base[16 + lhalf * 8 + 4];
      a[mi] = make_frag(q0, q1, q2, q3);
    }
    v16h bf[2];
    #pragma unroll
    for (int ni = 0; ni < 2; ++ni) {
      const float* base = &fB[cur][(wn * 32 + ni * 16 + lmod) * LDPF];
      const float4 q0 = *(const float4*)&base[lhalf * 8];
      const float4 q1 = *(const float4*)&base[lhalf * 8 + 4];
      const float4 q2 = *(const float4*)&base[16 + lhalf * 8];
      const float4 q3 = *(const float4*)&base[16 + lhalf * 8 + 4];
      bf[ni] = make_frag(q0, q1, q2, q3);
    }
    __builtin_amdgcn_sched_barrier(0);     // all 6 frags live before WMMA issue

    // (3) 8 back-to-back WMMAs
    #pragma unroll
    for (int mi = 0; mi < 4; ++mi)
      #pragma unroll
      for (int ni = 0; ni < 2; ++ni)
        acc[mi][ni] = __builtin_amdgcn_wmma_f32_16x16x32_f16(
            false, a[mi], false, bf[ni], (short)0, acc[mi][ni], false, false);
    __builtin_amdgcn_sched_barrier(0);

    // (4) drain this wave's async copies, then block-wide barrier
    if (more) wait_async0();
    __syncthreads();
  }

  // Epilogue: scatter-add (each token gets exactly 2 expert contributions;
  // two-term fp32 atomic adds are commutative -> deterministic output)
  #pragma unroll
  for (int mi = 0; mi < 4; ++mi) {
    #pragma unroll
    for (int ni = 0; ni < 2; ++ni) {
      const int nG = n0 + wn * 32 + ni * 16 + lmod;
      #pragma unroll
      for (int rr = 0; rr < 8; ++rr) {
        const int mrow = wm * 64 + mi * 16 + lhalf * 8 + rr;  // C layout: M = r + lhalf*8
        if (m0 + mrow < count) {
          atomicAdd(out + (size_t)sTok[mrow] * DM + nG, acc[mi][ni][rr]);
        }
      }
    }
  }
}

// ---------------------------------------------------------------------------
extern "C" void kernel_launch(void* const* d_in, const int* in_sizes, int n_in,
                              void* d_out, int out_size, void* d_ws, size_t ws_size,
                              hipStream_t stream) {
  const float* H  = (const float*)d_in[0];  // [T, D]
  const float* Wg = (const float*)d_in[1];  // [E, D]
  const float* bg = (const float*)d_in[2];  // [E]
  const float* W  = (const float*)d_in[3];  // [E, D, D]
  const float* b  = (const float*)d_in[4];  // [E, D]
  float* out = (float*)d_out;               // [T, D]

  int* cnt   = (int*)d_ws;                        // 8 counters
  int* lists = (int*)((char*)d_ws + 256);         // 8 * T token ids (512 KB)

  hipMemsetAsync(cnt, 0, 256, stream);
  hipMemsetAsync(out, 0, sizeof(float) * (size_t)T_TOKENS * DM, stream);

  moe_router_kernel<<<512, 256, 0, stream>>>(H, Wg, bg, cnt, lists);

  dim3 grid(T_TOKENS / BM, DM / BN, NEXP);        // inactive tiles exit early
  moe_gemm_kernel<<<grid, 256, 0, stream>>>(H, W, b, out, cnt, lists);
}
